// StructureAwareTransformerLayer_26362509263402
// MI455X (gfx1250) — matrix operand order repaired and checked by reading, not compiled, not run on hardware
//
#include <hip/hip_runtime.h>
#include <hip/hip_bf16.h>
#include <math.h>
#include <stdint.h>

// ---------------------------------------------------------------- types
typedef __bf16 bf16_t;
typedef __attribute__((ext_vector_type(16))) __bf16 v16bf;
typedef __attribute__((ext_vector_type(8)))  __bf16 v8bf;
typedef __attribute__((ext_vector_type(8)))  float  v8f;

#define B_  4
#define S_  2048
#define D_  512
#define H_  8
#define HD_ 64
#define N_  1024
#define E_  16384
#define M_  (B_ * S_)      // 8192 rows for token-major tensors
#define MN_ (B_ * N_)      // 4096 rows for node-major tensors

// low 32 bits of a generic pointer to __shared__ = LDS byte offset (ISA 10.2)
__device__ __forceinline__ uint32_t lds_off(const void* p) {
  return (uint32_t)(size_t)p;
}

// =====================================================================
// Generic GEMM: C[M,N] = act( A_f32[M,K] @ Bt_bf16[N,K] + bias + residual )
// A converted f32->bf16 while staging into LDS (batched loads);
// B tile copied global->LDS with GLOBAL_LOAD_ASYNC_TO_LDS_B128 (ASYNCcnt).
// BM=128, BN=128, BK=32; 8 waves (4x2); each wave -> 32x64 (2x4 WMMA tiles).
// Requires M%128==0, N%128==0, K%32==0 (true for all call sites).
// act: 0 none, 1 exact GELU, 2 sigmoid, 3 relu
// =====================================================================
__global__ __launch_bounds__(256) void k_gemm(
    const float* __restrict__ A, const bf16_t* __restrict__ Bt,
    const float* __restrict__ bias, const float* __restrict__ res,
    float* __restrict__ C, int M, int N, int K, int act)
{
  __shared__ bf16_t As[128][40];
  __shared__ bf16_t Bs[128][40];
  const int tid = threadIdx.x;
  const int lane = tid & 31, w = tid >> 5;
  const int wm = w >> 1, wn = w & 1;
  const int hlf = lane >> 4, l16 = lane & 15;
  const int m0 = blockIdx.x * 128, n0 = blockIdx.y * 128;

  const int achunk = (tid & 7) * 4;   // 4 floats per chunk, 8 chunks/row
  const int arow   = tid >> 3;        // 32 rows per pass, 4 passes
  const int brow   = tid >> 1;        // 128 rows, 32 bf16 each
  const int bcol   = (tid & 1) * 16;  // two 16-byte async chunks per thread

  v8f acc[2][4] = {};

  for (int kt = 0; kt < K; kt += 32) {
    // ---- async copy B tile (bf16 -> bf16, no VGPR round trip) ----
    {
      const uint32_t l0 = lds_off(&Bs[brow][bcol]);
      const uint32_t g0 = (uint32_t)(((size_t)(n0 + brow) * K + kt + bcol) * sizeof(bf16_t));
      asm volatile(
        "global_load_async_to_lds_b128 %0, %1, %2\n\t"
        "global_load_async_to_lds_b128 %3, %4, %2"
        :: "v"(l0), "v"(g0), "s"(Bt), "v"(l0 + 16u), "v"(g0 + 16u)
        : "memory");
    }
    // ---- stage A tile (128x32): batch loads, then convert ----
    float4 fa[4];
    #pragma unroll
    for (int p = 0; p < 4; ++p)
      fa[p] = *(const float4*)(A + (size_t)(m0 + arow + p * 32) * K + kt + achunk);
    #pragma unroll
    for (int p = 0; p < 4; ++p) {
      bf16_t* d = &As[arow + p * 32][achunk];
      d[0] = (bf16_t)fa[p].x; d[1] = (bf16_t)fa[p].y;
      d[2] = (bf16_t)fa[p].z; d[3] = (bf16_t)fa[p].w;
    }
    asm volatile("s_wait_asynccnt 0x0" ::: "memory");
    __syncthreads();

    v16bf af[2], bf[4];
    #pragma unroll
    for (int i = 0; i < 2; ++i) {
      const bf16_t* pa = &As[wm * 32 + i * 16 + l16][hlf * 8];
      v8bf lo = *(const v8bf*)pa;
      v8bf hi = *(const v8bf*)(pa + 16);
      #pragma unroll
      for (int e = 0; e < 8; ++e) { af[i][e] = lo[e]; af[i][e + 8] = hi[e]; }
    }
    #pragma unroll
    for (int j = 0; j < 4; ++j) {
      const bf16_t* pb = &Bs[wn * 64 + j * 16 + l16][hlf * 8];
      v8bf lo = *(const v8bf*)pb;
      v8bf hi = *(const v8bf*)(pb + 16);
      #pragma unroll
      for (int e = 0; e < 8; ++e) { bf[j][e] = lo[e]; bf[j][e + 8] = hi[e]; }
    }
    #pragma unroll
    for (int i = 0; i < 2; ++i)
      #pragma unroll
      for (int j = 0; j < 4; ++j)
        acc[i][j] = __builtin_amdgcn_wmma_f32_16x16x32_bf16(
            false, af[i], false, bf[j], (short)0, acc[i][j], false, false);
    __syncthreads();
  }

  // ---- epilogue ----
  const int mbase = m0 + wm * 32, nbase = n0 + wn * 64;
  #pragma unroll
  for (int i = 0; i < 2; ++i) {
    #pragma unroll
    for (int j = 0; j < 4; ++j) {
      const int n = nbase + j * 16 + l16;
      const float bv = bias ? bias[n] : 0.0f;
      #pragma unroll
      for (int r = 0; r < 8; ++r) {
        const int m = mbase + i * 16 + hlf * 8 + r;
        float v = acc[i][j][r] + bv;
        if (res) v += res[(size_t)m * N + n];
        if (act == 1)      v = 0.5f * v * (1.0f + erff(v * 0.70710678118654752f));
        else if (act == 2) v = 1.0f / (1.0f + __expf(-v));
        else if (act == 3) v = fmaxf(v, 0.0f);
        C[(size_t)m * N + n] = v;
      }
    }
  }
}

// =====================================================================
// Flash-style attention. Q,K,V bf16 [B*H, S, 64]; O f32 [B, S, 512].
// Block: 128 threads (4 waves); block covers 64 query rows of one (b,h);
// each wave owns 16 query rows. Key blocks of 64 staged in LDS:
// K tile via async global->LDS copy; V transposed through VGPRs.
// =====================================================================
__global__ __launch_bounds__(128) void k_attn(
    const bf16_t* __restrict__ Q, const bf16_t* __restrict__ Kc,
    const bf16_t* __restrict__ V, float* __restrict__ O)
{
  __shared__ bf16_t Ks[64][64];       // [key][hd]
  __shared__ bf16_t Vt[64][64];       // [hd][key] (transposed)
  __shared__ bf16_t Pw[4][16][64];    // per-wave probabilities [m][key]

  const int tid = threadIdx.x, lane = tid & 31, wq = tid >> 5;
  const int hlf = lane >> 4, l16 = lane & 15;
  const int bh = blockIdx.y;          // 0..B*H-1
  const int b = bh >> 3, h = bh & 7;
  const int q0 = blockIdx.x * 64 + wq * 16;

  const bf16_t* Qb = Q + (size_t)bh * S_ * HD_;
  const bf16_t* Kb = Kc + (size_t)bh * S_ * HD_;
  const bf16_t* Vb = V + (size_t)bh * S_ * HD_;

  // Q fragments for the two 32-wide K-steps over hd
  v16bf aq[2];
  #pragma unroll
  for (int ks = 0; ks < 2; ++ks) {
    const bf16_t* p = Qb + (size_t)(q0 + l16) * HD_ + ks * 32 + hlf * 8;
    v8bf lo = *(const v8bf*)p, hi = *(const v8bf*)(p + 16);
    #pragma unroll
    for (int e = 0; e < 8; ++e) { aq[ks][e] = lo[e]; aq[ks][e + 8] = hi[e]; }
  }

  v8f accO[4] = {};
  float mrun[8], lrun[8];
  #pragma unroll
  for (int r = 0; r < 8; ++r) { mrun[r] = -1e30f; lrun[r] = 0.0f; }

  for (int kb = 0; kb < S_; kb += 64) {
    // K tile: async global->LDS copy; V tile: load + transpose store
    #pragma unroll
    for (int c = 0; c < 4; ++c) {
      const int idx = tid + c * 128;             // 512 chunks of 8 bf16
      const int row = idx >> 3, col = (idx & 7) * 8;
      const uint32_t lk = lds_off(&Ks[row][col]);
      const uint32_t gk = (uint32_t)(((size_t)(kb + row) * HD_ + col) * sizeof(bf16_t));
      asm volatile("global_load_async_to_lds_b128 %0, %1, %2"
                   :: "v"(lk), "v"(gk), "s"(Kb) : "memory");
      v8bf tv = *(const v8bf*)(Vb + (size_t)(kb + row) * HD_ + col);
      #pragma unroll
      for (int e = 0; e < 8; ++e) Vt[col + e][row] = tv[e];
    }
    asm volatile("s_wait_asynccnt 0x0" ::: "memory");
    __syncthreads();

    // scores: 16 q rows x 64 keys
    v8f sc[4];
    #pragma unroll
    for (int nt = 0; nt < 4; ++nt) {
      v8f a = {};
      #pragma unroll
      for (int ks = 0; ks < 2; ++ks) {
        v16bf bf;
        const bf16_t* pb = &Ks[nt * 16 + l16][ks * 32 + hlf * 8];
        v8bf lo = *(const v8bf*)pb, hi = *(const v8bf*)(pb + 16);
        #pragma unroll
        for (int e = 0; e < 8; ++e) { bf[e] = lo[e]; bf[e + 8] = hi[e]; }
        a = __builtin_amdgcn_wmma_f32_16x16x32_bf16(
            false, aq[ks], false, bf, (short)0, a, false, false);
      }
      sc[nt] = a * 0.125f;   // 1/sqrt(64)
    }

    // online softmax: row max across 64 keys (4 tiles + 16-lane group reduce)
    float mloc[8];
    #pragma unroll
    for (int r = 0; r < 8; ++r)
      mloc[r] = fmaxf(fmaxf(sc[0][r], sc[1][r]), fmaxf(sc[2][r], sc[3][r]));
    #pragma unroll
    for (int msk = 1; msk < 16; msk <<= 1)
      #pragma unroll
      for (int r = 0; r < 8; ++r)
        mloc[r] = fmaxf(mloc[r], __shfl_xor(mloc[r], msk, 32));

    float corr[8], mnew[8];
    #pragma unroll
    for (int r = 0; r < 8; ++r) {
      mnew[r] = fmaxf(mrun[r], mloc[r]);
      corr[r] = __expf(mrun[r] - mnew[r]);
      mrun[r] = mnew[r];
      lrun[r] *= corr[r];
    }
    #pragma unroll
    for (int j = 0; j < 4; ++j)
      #pragma unroll
      for (int r = 0; r < 8; ++r) accO[j][r] *= corr[r];

    float lloc[8] = {};
    #pragma unroll
    for (int nt = 0; nt < 4; ++nt)
      #pragma unroll
      for (int r = 0; r < 8; ++r) {
        const float p = __expf(sc[nt][r] - mnew[r]);
        sc[nt][r] = p;
        lloc[r] += p;
      }
    #pragma unroll
    for (int msk = 1; msk < 16; msk <<= 1)
      #pragma unroll
      for (int r = 0; r < 8; ++r) lloc[r] += __shfl_xor(lloc[r], msk, 32);
    #pragma unroll
    for (int r = 0; r < 8; ++r) lrun[r] += lloc[r];

    // C-fragment -> A-fragment reshape via per-wave LDS
    #pragma unroll
    for (int nt = 0; nt < 4; ++nt)
      #pragma unroll
      for (int r = 0; r < 8; ++r)
        Pw[wq][hlf * 8 + r][nt * 16 + l16] = (bf16_t)sc[nt][r];

    // accO += P @ V
    #pragma unroll
    for (int ks2 = 0; ks2 < 2; ++ks2) {
      v16bf ap;
      const bf16_t* pp = &Pw[wq][l16][ks2 * 32 + hlf * 8];
      v8bf lo = *(const v8bf*)pp, hi = *(const v8bf*)(pp + 16);
      #pragma unroll
      for (int e = 0; e < 8; ++e) { ap[e] = lo[e]; ap[e + 8] = hi[e]; }
      #pragma unroll
      for (int j = 0; j < 4; ++j) {
        v16bf bf;
        const bf16_t* pb = &Vt[j * 16 + l16][ks2 * 32 + hlf * 8];
        v8bf blo = *(const v8bf*)pb, bhi = *(const v8bf*)(pb + 16);
        #pragma unroll
        for (int e = 0; e < 8; ++e) { bf[e] = blo[e]; bf[e + 8] = bhi[e]; }
        accO[j] = __builtin_amdgcn_wmma_f32_16x16x32_bf16(
            false, ap, false, bf, (short)0, accO[j], false, false);
      }
    }
    __syncthreads();
  }

  // normalize + write [B,S,D] with head offset (merges transpose back)
  #pragma unroll
  for (int j = 0; j < 4; ++j)
    #pragma unroll
    for (int r = 0; r < 8; ++r) {
      const int m = q0 + hlf * 8 + r;
      const int n = h * HD_ + j * 16 + l16;
      O[((size_t)b * S_ + m) * D_ + n] = accO[j][r] / lrun[r];
    }
}

// ================================================ small helper kernels
__global__ __launch_bounds__(256) void k_ln(
    const float* __restrict__ X, const float* __restrict__ g,
    const float* __restrict__ be, float* __restrict__ Y)
{
  __shared__ float red[256];
  const int row = blockIdx.x;
  const float* xr = X + (size_t)row * D_;
  float s = 0.f;
  for (int d = threadIdx.x; d < D_; d += 256) s += xr[d];
  red[threadIdx.x] = s; __syncthreads();
  for (int o = 128; o > 0; o >>= 1) { if (threadIdx.x < o) red[threadIdx.x] += red[threadIdx.x + o]; __syncthreads(); }
  const float mean = red[0] * (1.0f / D_);
  __syncthreads();
  float v = 0.f;
  for (int d = threadIdx.x; d < D_; d += 256) { const float t = xr[d] - mean; v += t * t; }
  red[threadIdx.x] = v; __syncthreads();
  for (int o = 128; o > 0; o >>= 1) { if (threadIdx.x < o) red[threadIdx.x] += red[threadIdx.x + o]; __syncthreads(); }
  const float rstd = rsqrtf(red[0] * (1.0f / D_) + 1e-5f);
  float* yr = Y + (size_t)row * D_;
  for (int d = threadIdx.x; d < D_; d += 256) yr[d] = (xr[d] - mean) * rstd * g[d] + be[d];
}

// W f32 [K,N] -> Wt bf16 [N,K]
__global__ void k_pack_w(const float* __restrict__ W, bf16_t* __restrict__ Wt, int K, int N)
{
  const int idx = blockIdx.x * 256 + threadIdx.x;
  if (idx >= N * K) return;
  const int n = idx / K, k = idx % K;
  Wt[idx] = (bf16_t)W[(size_t)k * N + n];
}

// f32 [B,S,512] -> bf16 [B,8,S,64]
__global__ void k_split(const float* __restrict__ X, bf16_t* __restrict__ Y)
{
  const size_t idx = (size_t)blockIdx.x * 256 + threadIdx.x;
  if (idx >= (size_t)M_ * D_) return;
  const int d = (int)(idx % D_);
  const size_t bs = idx / D_;
  const int s = (int)(bs % S_), b = (int)(bs / S_);
  const int h = d >> 6, dd = d & 63;
  Y[(((size_t)(b * H_ + h)) * S_ + s) * HD_ + dd] = (bf16_t)X[idx];
}

__global__ void k_zero(float* __restrict__ p, size_t n)
{
  const size_t idx = (size_t)blockIdx.x * 256 + threadIdx.x;
  if (idx < n) p[idx] = 0.0f;
}

__global__ void k_gather(const float* __restrict__ nx2, const int* __restrict__ nidx,
                         float* __restrict__ nodes)
{
  const size_t idx = (size_t)blockIdx.x * 256 + threadIdx.x;
  if (idx >= (size_t)MN_ * D_) return;
  const int d = (int)(idx % D_);
  const size_t bn = idx / D_;
  const int n = (int)(bn % N_), b = (int)(bn / N_);
  const int s = nidx[b * N_ + n];
  nodes[idx] = nx2[((size_t)b * S_ + s) * D_ + d];
}

__global__ void k_edge(const float* __restrict__ nodes, const int* __restrict__ ei,
                       float* __restrict__ agg)
{
  const size_t idx = (size_t)blockIdx.x * 256 + threadIdx.x;
  if (idx >= (size_t)B_ * E_ * (D_ / 4)) return;
  const int dc = (int)(idx % (D_ / 4));
  const size_t be = idx / (D_ / 4);
  const int e = (int)(be % E_), b = (int)(be / E_);
  const int src = ei[e], dst = ei[E_ + e];
  const float* sp = nodes + ((size_t)b * N_ + src) * D_ + dc * 4;
  float* dp = agg + ((size_t)b * N_ + dst) * D_ + dc * 4;
  #pragma unroll
  for (int c = 0; c < 4; ++c) atomicAdd(dp + c, sp[c]);
}

__global__ void k_scatter(const float* __restrict__ upd, const int* __restrict__ nidx,
                          float* __restrict__ gnn)
{
  const size_t idx = (size_t)blockIdx.x * 256 + threadIdx.x;
  if (idx >= (size_t)MN_ * D_) return;
  const int d = (int)(idx % D_);
  const size_t bn = idx / D_;
  const int n = (int)(bn % N_), b = (int)(bn / N_);
  const int s = nidx[b * N_ + n];
  gnn[((size_t)b * S_ + s) * D_ + d] = upd[idx];
}

__global__ void k_concat(const float* __restrict__ a, const float* __restrict__ c,
                         float* __restrict__ out)
{
  const size_t idx = (size_t)blockIdx.x * 256 + threadIdx.x;
  if (idx >= (size_t)M_ * 2 * D_) return;
  const int k = (int)(idx % (2 * D_));
  const size_t m = idx / (2 * D_);
  out[idx] = (k < D_) ? a[m * D_ + k] : c[m * D_ + (k - D_)];
}

__global__ void k_gatecomb(const float* __restrict__ x1, const float* __restrict__ gate,
                           const float* __restrict__ gnn, const float* __restrict__ nx2,
                           float* __restrict__ x2)
{
  const size_t idx = (size_t)blockIdx.x * 256 + threadIdx.x;
  if (idx >= (size_t)M_ * D_) return;
  const float gv = gate[idx];
  x2[idx] = x1[idx] + gv * gnn[idx] + (1.0f - gv) * nx2[idx];
}

// ================================================================ host
static inline int gblk(size_t n) { return (int)((n + 255) / 256); }

extern "C" void kernel_launch(void* const* d_in, const int* in_sizes, int n_in,
                              void* d_out, int out_size, void* d_ws, size_t ws_size,
                              hipStream_t stream) {
  (void)in_sizes; (void)n_in; (void)out_size; (void)ws_size;
  const float* x   = (const float*)d_in[0];
  const int*   ei  = (const int*)d_in[1];
  const int*   nid = (const int*)d_in[2];
  const float* Wq  = (const float*)d_in[3];  const float* bq  = (const float*)d_in[4];
  const float* Wk  = (const float*)d_in[5];  const float* bk  = (const float*)d_in[6];
  const float* Wv  = (const float*)d_in[7];  const float* bv  = (const float*)d_in[8];
  const float* Wo  = (const float*)d_in[9];  const float* bo  = (const float*)d_in[10];
  const float* Wf1 = (const float*)d_in[11]; const float* bf1 = (const float*)d_in[12];
  const float* Wf2 = (const float*)d_in[13]; const float* bf2 = (const float*)d_in[14];
  const float* g1  = (const float*)d_in[15]; const float* be1 = (const float*)d_in[16];
  const float* g2  = (const float*)d_in[17]; const float* be2 = (const float*)d_in[18];
  const float* g3  = (const float*)d_in[19]; const float* be3 = (const float*)d_in[20];
  const float* Wg  = (const float*)d_in[21]; const float* bg  = (const float*)d_in[22];
  const float* Wgs = (const float*)d_in[23]; const float* Wgn = (const float*)d_in[24];
  const float* bgn = (const float*)d_in[25];

  char* ws = (char*)d_ws;
  const size_t SZ_LN = (size_t)M_ * D_ * 4;              // 16 MB token-major f32
  const size_t SZ_ND = (size_t)MN_ * D_ * 4;             // 8 MB node-major f32
  const size_t SZ_HB = (size_t)(B_ * H_) * S_ * HD_ * 2; // 8 MB per bf16 head tensor

  size_t off = 0;
  const size_t o_wp   = off; off += 4194304ull * 2;      // packed bf16 weights
  const size_t o_ln   = off; off += SZ_LN;               // nx -> nx2 -> nx3
  const size_t o_x1   = off; off += SZ_LN;
  const size_t o_x2   = off; off += SZ_LN;
  const size_t o_big  = off; off += (size_t)M_ * 2048 * 4; // qkv_f32 / cat+gate / ffn hidden
  const size_t o_qkvb = off; off += 3 * SZ_HB;           // q,k,v bf16 -> nodes/agg/tbuf
  const size_t o_aog  = off; off += SZ_LN;               // ao -> gnn_out

  bf16_t* wp  = (bf16_t*)(ws + o_wp);
  bf16_t* wqT = wp;            bf16_t* wkT = wp + 262144;  bf16_t* wvT = wp + 524288;
  bf16_t* woT = wp + 786432;   bf16_t* wgsT = wp + 1048576; bf16_t* wgnT = wp + 1310720;
  bf16_t* wgT = wp + 1572864;  bf16_t* wf1T = wp + 2097152; bf16_t* wf2T = wp + 3145728;

  float* nx   = (float*)(ws + o_ln);    // also nx2, nx3 (sequential reuse)
  float* x1   = (float*)(ws + o_x1);
  float* x2   = (float*)(ws + o_x2);
  float* qf   = (float*)(ws + o_big);
  float* kf   = qf + (size_t)M_ * D_;
  float* vf   = kf + (size_t)M_ * D_;
  float* cat  = (float*)(ws + o_big);                       // reuse after split
  float* gate = (float*)(ws + o_big + (size_t)M_ * 2 * D_ * 4 / 2); // +16 MB
  float* h1   = (float*)(ws + o_big);                       // reuse for FFN hidden
  bf16_t* qb  = (bf16_t*)(ws + o_qkvb);
  bf16_t* kb  = (bf16_t*)(ws + o_qkvb + SZ_HB);
  bf16_t* vb  = (bf16_t*)(ws + o_qkvb + 2 * SZ_HB);
  float* nodes = (float*)(ws + o_qkvb);                     // reuse after attention
  float* agg   = (float*)(ws + o_qkvb + SZ_ND);
  float* tbuf  = (float*)(ws + o_qkvb + 2 * SZ_ND);
  float* ao    = (float*)(ws + o_aog);
  float* gnn   = (float*)(ws + o_aog);                      // reuse after x1
  float* out   = (float*)d_out;

  // ---- pack all weights to bf16 [N][K] ----
  k_pack_w<<<gblk(262144),  256, 0, stream>>>(Wq,  wqT,  512, 512);
  k_pack_w<<<gblk(262144),  256, 0, stream>>>(Wk,  wkT,  512, 512);
  k_pack_w<<<gblk(262144),  256, 0, stream>>>(Wv,  wvT,  512, 512);
  k_pack_w<<<gblk(262144),  256, 0, stream>>>(Wo,  woT,  512, 512);
  k_pack_w<<<gblk(262144),  256, 0, stream>>>(Wgs, wgsT, 512, 512);
  k_pack_w<<<gblk(262144),  256, 0, stream>>>(Wgn, wgnT, 512, 512);
  k_pack_w<<<gblk(524288),  256, 0, stream>>>(Wg,  wgT,  1024, 512);
  k_pack_w<<<gblk(1048576), 256, 0, stream>>>(Wf1, wf1T, 512, 2048);
  k_pack_w<<<gblk(1048576), 256, 0, stream>>>(Wf2, wf2T, 2048, 512);

  // ---- stage A: LN1 ----
  k_ln<<<M_, 256, 0, stream>>>(x, g1, be1, nx);

  // ---- stage B: QKV projections ----
  dim3 gq(M_ / 128, 512 / 128);
  k_gemm<<<gq, 256, 0, stream>>>(nx, wqT, bq, nullptr, qf, M_, 512, 512, 0);
  k_gemm<<<gq, 256, 0, stream>>>(nx, wkT, bk, nullptr, kf, M_, 512, 512, 0);
  k_gemm<<<gq, 256, 0, stream>>>(nx, wvT, bv, nullptr, vf, M_, 512, 512, 0);
  k_split<<<gblk((size_t)M_ * D_), 256, 0, stream>>>(qf, qb);
  k_split<<<gblk((size_t)M_ * D_), 256, 0, stream>>>(kf, kb);
  k_split<<<gblk((size_t)M_ * D_), 256, 0, stream>>>(vf, vb);

  // ---- stage C: flash attention ----
  k_attn<<<dim3(S_ / 64, B_ * H_), 128, 0, stream>>>(qb, kb, vb, ao);

  // ---- stage D: output projection + residual ----
  k_gemm<<<gq, 256, 0, stream>>>(ao, woT, bo, x, x1, M_, 512, 512, 0);
  k_ln<<<M_, 256, 0, stream>>>(x1, g2, be2, nx);   // nx now holds nx2

  // ---- stage E: GNN + gated fusion ----
  k_gather<<<gblk((size_t)MN_ * D_), 256, 0, stream>>>(nx, nid, nodes);
  k_zero<<<gblk((size_t)MN_ * D_), 256, 0, stream>>>(agg, (size_t)MN_ * D_);
  k_edge<<<gblk((size_t)B_ * E_ * (D_ / 4)), 256, 0, stream>>>(nodes, ei, agg);
  dim3 gn(MN_ / 128, 512 / 128);
  k_gemm<<<gn, 256, 0, stream>>>(nodes, wgsT, nullptr, nullptr, tbuf, MN_, 512, 512, 0);
  k_gemm<<<gn, 256, 0, stream>>>(agg, wgnT, bgn, tbuf, tbuf, MN_, 512, 512, 3); // upd=relu
  k_zero<<<gblk((size_t)M_ * D_), 256, 0, stream>>>(gnn, (size_t)M_ * D_);
  k_scatter<<<gblk((size_t)MN_ * D_), 256, 0, stream>>>(tbuf, nid, gnn);
  k_concat<<<gblk((size_t)M_ * 2 * D_), 256, 0, stream>>>(nx, gnn, cat);
  k_gemm<<<gq, 256, 0, stream>>>(cat, wgT, bg, nullptr, gate, M_, 512, 1024, 2); // sigmoid
  k_gatecomb<<<gblk((size_t)M_ * D_), 256, 0, stream>>>(x1, gate, gnn, nx, x2);

  // ---- stage F: FFN ----
  k_ln<<<M_, 256, 0, stream>>>(x2, g3, be3, nx);   // nx now holds nx3
  dim3 gf1(M_ / 128, 2048 / 128);
  k_gemm<<<gf1, 256, 0, stream>>>(nx, wf1T, bf1, nullptr, h1, M_, 2048, 512, 1); // gelu
  k_gemm<<<gq, 256, 0, stream>>>(h1, wf2T, bf2, x2, out, M_, 512, 2048, 0);
}
